// ProbMultiHeadedAttention_88699664597587
// MI455X (gfx1250) — matrix operand order repaired and checked by reading, not compiled, not run on hardware
//
#include <hip/hip_runtime.h>

// ---------------------------------------------------------------------------
// ProbSparse multi-head attention (Informer-style) for MI455X / gfx1250.
// B=16, L=1024, F=512, H=8, D=64, u=35 (padded to 48 = 3 row-tiles of 16).
// Heavy math: v_wmma_f32_16x16x32_bf16 (wave32). ctx tiles staged via the
// Tensor Data Mover (tensor_load_to_lds + s_wait_tensorcnt).
// ---------------------------------------------------------------------------

#define B_  16
#define L_  1024
#define F_  512
#define H_  8
#define D_  64
#define U_  35
#define UP_ 48          // padded top-k rows (3 tiles of 16)
#define BH_ (B_ * H_)   // 128

typedef __attribute__((ext_vector_type(16))) __bf16 bf16x16;
typedef __attribute__((ext_vector_type(8)))  __bf16 bf16x8;
typedef __attribute__((ext_vector_type(4)))  __bf16 bf16x4;
typedef __attribute__((ext_vector_type(8)))  float  v8f;
typedef __attribute__((ext_vector_type(4)))  unsigned int u32x4;
typedef __attribute__((ext_vector_type(8)))  int i32x8;
typedef __attribute__((ext_vector_type(4)))  int i32x4;

// --------------------------- WMMA helpers ----------------------------------

__device__ __forceinline__ v8f wmma_bf16(bf16x16 a, bf16x16 b, v8f c) {
    return __builtin_amdgcn_wmma_f32_16x16x32_bf16(
        /*neg_a=*/false, a, /*neg_b=*/false, b,
        /*c_mod=*/(short)0, c, /*reuse_a=*/false, /*reuse_b=*/false);
}

__device__ __forceinline__ bf16x16 frag_pack(const __bf16* p0, const __bf16* p1) {
    bf16x8 lo = *(const bf16x8*)p0;
    bf16x8 hi = *(const bf16x8*)p1;
    bf16x16 r;
#pragma unroll
    for (int i = 0; i < 8; ++i) { r[i] = lo[i]; r[i + 8] = hi[i]; }
    return r;
}

// A-matrix 16x32 bf16 fragment (ISA 7.12.2): lane<16 -> M=lane, K chunks {0..7,16..23};
// lane>=16 -> M=lane-16, K chunks {8..15,24..31}.
__device__ __forceinline__ bf16x16 frag_a(const __bf16* base, int pitch) {
    int lane = threadIdx.x & 31;
    int row  = lane & 15;
    int kb   = (lane & 16) ? 8 : 0;
    const __bf16* p = base + row * pitch + kb;
    return frag_pack(p, p + 16);
}

// B-matrix 32x16 bf16 fragment: lane<16 -> N=lane, K {0..7,16..23};
// lane>=16 -> N=lane-16, K {8..15,24..31} (base must be [k][n] layout).
__device__ __forceinline__ bf16x16 frag_b(const __bf16* base, int pitch) {
    int lane = threadIdx.x & 31;
    int n    = lane & 15;
    int kb   = (lane & 16) ? 8 : 0;
    bf16x16 r;
#pragma unroll
    for (int i = 0; i < 8; ++i) {
        r[i]     = base[(kb + i) * pitch + n];
        r[i + 8] = base[(16 + kb + i) * pitch + n];
    }
    return r;
}

// ---------------------- Tensor Data Mover (TDM) helper ---------------------
// 2-D tile load: global (row-major, strideElems between rows, 2-byte elems)
// -> LDS packed rows. Descriptor per CDNA5 ISA §8.3/8.4.
__device__ __forceinline__ void tdm_load_2d_bf16(unsigned ldsOff, const void* gptr,
                                                 unsigned tileX, unsigned tileY,
                                                 unsigned strideElems,
                                                 unsigned dimX, unsigned dimY)
{
    unsigned long long ga = (unsigned long long)(uintptr_t)gptr;
    u32x4 g0;
    g0[0] = 1u;                                        // count=1, user descriptor
    g0[1] = ldsOff;                                    // lds_addr
    g0[2] = (unsigned)(ga & 0xFFFFFFFFu);              // global_addr[31:0]
    g0[3] = (unsigned)((ga >> 32) & 0x01FFFFFFu) | (2u << 30);  // addr[56:32] | type=2
    i32x8 g1;
    g1[0] = (int)(1u << 16);            // workgroup_mask=0, data_size=1 (2 bytes)
    g1[1] = (int)(dimX << 16);          // [63:48] tensor_dim0[15:0]
    g1[2] = (int)(dimY << 16);          // [79:64] dim0 hi=0, [95:80] tensor_dim1[15:0]
    g1[3] = (int)(tileX << 16);         // [111:96] dim1 hi=0, [127:112] tile_dim0
    g1[4] = (int)(tileY & 0xFFFFu);     // [143:128] tile_dim1, [159:144] tile_dim2=0
    g1[5] = (int)strideElems;           // [191:160] tensor_dim0_stride[31:0]
    g1[6] = 0;                          // stride hi, dim1_stride lo
    g1[7] = 0;
    i32x4 z4 = {};
#if defined(__clang_major__) && __clang_major__ >= 23
    i32x8 z8 = {};
    __builtin_amdgcn_tensor_load_to_lds(g0, g1, z4, z4, z8, 0);
#else
    __builtin_amdgcn_tensor_load_to_lds(g0, g1, z4, z4, 0);
#endif
}

__device__ __forceinline__ unsigned lds_offset(const void* p) {
    // generic LDS address: low 32 bits are the wave-relative LDS offset
    return (unsigned)(uintptr_t)p;
}

// --------------------------- Kernel 1: X @ W^T + b -------------------------
// out[m,n] = sum_k X[m,k] * W[n,k] + b[n].  X f32 (M x 512), W f32 (512 x 512).
// MODE 0: bf16 out in (B,H,L,D). MODE 1: f32 flat (M x 512).
#define GTM 128
#define GTN 64
#define GTK 32

template <int MODE>
__global__ void __launch_bounds__(256)
gemm_xwT_kernel(const float* __restrict__ X, const float* __restrict__ W,
                const float* __restrict__ bias,
                __bf16* __restrict__ outB, float* __restrict__ outF)
{
    __shared__ __bf16 ldsA[GTM][GTK];   // 8 KB
    __shared__ __bf16 ldsB[GTK][GTN];   // 4 KB, W tile transposed -> [k][n]

    const int t    = threadIdx.x;
    const int wave = t >> 5;
    const int lane = t & 31;
    const int m0   = blockIdx.x * GTM;
    const int n0   = blockIdx.y * GTN;
    const int wm   = (wave >> 1) * 32;
    const int wn   = (wave & 1) * 32;

    v8f acc[2][2] = {};

    for (int kk = 0; kk < F_; kk += GTK) {
        __syncthreads();
        {   // stage A: pack f32->bf16x8, two ds_store_b128 per thread
            int row = t >> 1, cb = (t & 1) * 16;
            const float* src = X + (size_t)(m0 + row) * F_ + kk + cb;
            __builtin_prefetch(src + GTK, 0, 1);
            bf16x8 v0, v1;
#pragma unroll
            for (int i = 0; i < 8; ++i) { v0[i] = (__bf16)src[i]; v1[i] = (__bf16)src[8 + i]; }
            *(bf16x8*)&ldsA[row][cb]     = v0;
            *(bf16x8*)&ldsA[row][cb + 8] = v1;
        }
        {   // stage B transposed: 64 n-rows x 32 k, 4 threads/n, 8 k each
            int n = t >> 2, kb = (t & 3) * 8;
            const float* src = W + (size_t)(n0 + n) * F_ + kk + kb;
#pragma unroll
            for (int i = 0; i < 8; ++i) ldsB[kb + i][n] = (__bf16)src[i];
        }
        __syncthreads();

        bf16x16 a0 = frag_a(&ldsA[wm][0],      GTK);
        bf16x16 a1 = frag_a(&ldsA[wm + 16][0], GTK);
        bf16x16 b0 = frag_b(&ldsB[0][wn],      GTN);
        bf16x16 b1 = frag_b(&ldsB[0][wn + 16], GTN);
        acc[0][0] = wmma_bf16(a0, b0, acc[0][0]);
        acc[0][1] = wmma_bf16(a0, b1, acc[0][1]);
        acc[1][0] = wmma_bf16(a1, b0, acc[1][0]);
        acc[1][1] = wmma_bf16(a1, b1, acc[1][1]);
    }

    const int n  = lane & 15;
    const int mb = (lane & 16) ? 8 : 0;
#pragma unroll
    for (int i = 0; i < 2; ++i)
#pragma unroll
        for (int j = 0; j < 2; ++j) {
            int gmBase = m0 + wm + i * 16 + mb;
            int gn     = n0 + wn + j * 16 + n;
            float bv   = bias[gn];
#pragma unroll
            for (int r = 0; r < 8; ++r) {
                int m = gmBase + r;
                float v = acc[i][j][r] + bv;
                if (MODE == 0) {
                    int b = m >> 10, l = m & 1023, h = gn >> 6, d = gn & 63;
                    outB[((((size_t)b * H_) + h) * L_ + l) * D_ + d] = (__bf16)v;
                } else {
                    outF[(size_t)m * F_ + gn] = v;
                }
            }
        }
}

// ------------------ Kernel 2: sparsity measure M = max - mean --------------
__global__ void __launch_bounds__(256)
scores_m_kernel(const __bf16* __restrict__ Qb, const __bf16* __restrict__ Kb,
                float* __restrict__ Mout)
{
    __shared__ __bf16 ldsQ[16][64];     // 2 KB
    __shared__ __bf16 ldsKT[64][128];   // 16 KB, [d][col]
    __shared__ float  wmax[8][16];
    __shared__ float  wsum[8][16];

    const int bh = blockIdx.x;
    const int q0 = blockIdx.y * 16;
    const int t = threadIdx.x, wave = t >> 5, lane = t & 31;

    {   // stage Q tile (16 x 64): 8-byte vector copies
        int r = t >> 4, cb = (t & 15) * 4;
        const __bf16* src = Qb + ((size_t)bh * L_ + q0 + r) * D_ + cb;
        *(bf16x4*)&ldsQ[r][cb] = *(const bf16x4*)src;
    }
    if (lane < 16) { wmax[wave][lane] = -3e38f; wsum[wave][lane] = 0.f; }
    __syncthreads();

    bf16x16 a0 = frag_a(&ldsQ[0][0],  64);
    bf16x16 a1 = frag_a(&ldsQ[0][32], 64);

    for (int kb = 0; kb < 8; ++kb) {
        __syncthreads();
        {   // stage K^T block: [d][c], vector global reads then b16 scatter
            int c = t >> 1, db = (t & 1) * 32;
            const bf16x8* src = (const bf16x8*)(Kb + ((size_t)bh * L_ + kb * 128 + c) * D_ + db);
#pragma unroll
            for (int q = 0; q < 4; ++q) {
                bf16x8 vv = src[q];
#pragma unroll
                for (int i = 0; i < 8; ++i) ldsKT[db + q * 8 + i][c] = vv[i];
            }
        }
        __syncthreads();

        const int c0 = wave * 16;
        v8f acc = {};
        acc = wmma_bf16(a0, frag_b(&ldsKT[0][c0],  128), acc);
        acc = wmma_bf16(a1, frag_b(&ldsKT[32][c0], 128), acc);

#pragma unroll
        for (int r = 0; r < 8; ++r) {
            float vmax = acc[r], vsum = acc[r];
#pragma unroll
            for (int m = 1; m < 16; m <<= 1) {
                vmax = fmaxf(vmax, __shfl_xor(vmax, m, 32));
                vsum += __shfl_xor(vsum, m, 32);
            }
            if ((lane & 15) == 0) {
                int row = ((lane & 16) ? 8 : 0) + r;
                wmax[wave][row] = fmaxf(wmax[wave][row], vmax);
                wsum[wave][row] += vsum;
            }
        }
    }
    __syncthreads();
    if (t < 16) {
        float mx = -3e38f, sm = 0.f;
#pragma unroll
        for (int w = 0; w < 8; ++w) { mx = fmaxf(mx, wmax[w][t]); sm += wsum[w][t]; }
        Mout[(size_t)bh * L_ + q0 + t] = mx - sm * (1.0f / (float)L_);
    }
}

// ------------------------ Kernel 3: top-35 indices -------------------------
__global__ void __launch_bounds__(256)
topk_kernel(const float* __restrict__ Mws, int* __restrict__ idxws)
{
    __shared__ float vals[L_];
    __shared__ float rv[256];
    __shared__ int   ri[256];
    const int bh = blockIdx.x, t = threadIdx.x;
    for (int i = t; i < L_; i += 256) vals[i] = Mws[(size_t)bh * L_ + i];
    __syncthreads();
    for (int it = 0; it < U_; ++it) {
        float bv = -3e38f; int bi = 0;
        for (int i = t; i < L_; i += 256) {
            float v = vals[i];
            if (v > bv) { bv = v; bi = i; }
        }
        rv[t] = bv; ri[t] = bi;
        __syncthreads();
        for (int s = 128; s > 0; s >>= 1) {
            if (t < s && rv[t + s] > rv[t]) { rv[t] = rv[t + s]; ri[t] = ri[t + s]; }
            __syncthreads();
        }
        if (t == 0) { idxws[bh * U_ + it] = ri[0]; vals[ri[0]] = -3e38f; }
        __syncthreads();
    }
}

// --------------- Kernel 4: reduced-Q scores (scaled + masked) --------------
__global__ void __launch_bounds__(256)
red_scores_kernel(const __bf16* __restrict__ Qb, const __bf16* __restrict__ Kb,
                  const int* __restrict__ idxws, const int* __restrict__ maskp,
                  float* __restrict__ Sws)
{
    __shared__ __bf16 ldsQ[16][64];
    __shared__ __bf16 ldsKT[64][128];

    const int bh = blockIdx.x, band = blockIdx.y;
    const int b  = bh >> 3;
    const int t = threadIdx.x, wave = t >> 5, lane = t & 31;

    {   // stage gathered Q rows
        int r = t >> 4, cb = (t & 15) * 4;
        int u = band * 16 + r;
        int row = (u < U_) ? idxws[bh * U_ + u] : 0;
        const __bf16* src = Qb + ((size_t)bh * L_ + row) * D_ + cb;
        *(bf16x4*)&ldsQ[r][cb] = *(const bf16x4*)src;
    }
    __syncthreads();

    bf16x16 a0 = frag_a(&ldsQ[0][0],  64);
    bf16x16 a1 = frag_a(&ldsQ[0][32], 64);
    const float scale = 0.125f;   // 1/sqrt(64)

    for (int kb = 0; kb < 8; ++kb) {
        __syncthreads();
        {
            int c = t >> 1, db = (t & 1) * 32;
            const bf16x8* src = (const bf16x8*)(Kb + ((size_t)bh * L_ + kb * 128 + c) * D_ + db);
#pragma unroll
            for (int q = 0; q < 4; ++q) {
                bf16x8 vv = src[q];
#pragma unroll
                for (int i = 0; i < 8; ++i) ldsKT[db + q * 8 + i][c] = vv[i];
            }
        }
        __syncthreads();

        const int c0 = wave * 16;
        v8f acc = {};
        acc = wmma_bf16(a0, frag_b(&ldsKT[0][c0],  128), acc);
        acc = wmma_bf16(a1, frag_b(&ldsKT[32][c0], 128), acc);

        const int n  = lane & 15;
        const int mb = (lane & 16) ? 8 : 0;
        const int cg = kb * 128 + c0 + n;
        const bool dead = (maskp[(size_t)b * L_ + cg] == 0);
#pragma unroll
        for (int r = 0; r < 8; ++r) {
            float v = dead ? -1e30f : acc[r] * scale;
            Sws[((size_t)bh * UP_ + band * 16 + mb + r) * L_ + cg] = v;
        }
    }
}

// ------------------------- Kernel 5: row softmax ---------------------------
__global__ void __launch_bounds__(256)
softmax_kernel(const float* __restrict__ Sws, __bf16* __restrict__ Pws)
{
    __shared__ float red[256];
    const int row = blockIdx.x, t = threadIdx.x;
    const float* S = Sws + (size_t)row * L_;
    __bf16* P = Pws + (size_t)row * L_;

    float mx = -3e38f;
    for (int i = t; i < L_; i += 256) mx = fmaxf(mx, S[i]);
    red[t] = mx; __syncthreads();
    for (int s = 128; s > 0; s >>= 1) { if (t < s) red[t] = fmaxf(red[t], red[t + s]); __syncthreads(); }
    mx = red[0]; __syncthreads();

    float sm = 0.f;
    for (int i = t; i < L_; i += 256) sm += __expf(S[i] - mx);
    red[t] = sm; __syncthreads();
    for (int s = 128; s > 0; s >>= 1) { if (t < s) red[t] += red[t + s]; __syncthreads(); }
    sm = red[0];
    float inv = (sm > 0.f) ? 1.f / sm : 0.f;
    for (int i = t; i < L_; i += 256) P[i] = (__bf16)(__expf(S[i] - mx) * inv);
}

// ------------------------ Kernel 6: ctx_top = P @ V ------------------------
// grid BH, 128 threads (4 waves); wave owns 16 d-columns, 3 row bands.
// Both LDS tiles are straight 2-D bf16 copies -> staged by the TDM:
// wave0 issues the V-tile descriptor, wave1 the P-tile descriptor.
__global__ void __launch_bounds__(128)
ctx_kernel(const __bf16* __restrict__ Pws, const __bf16* __restrict__ Vb,
           float* __restrict__ ctxtop)
{
    __shared__ __bf16 ldsP[UP_][32];   // 3 KB, rows packed (tile_dim0=32)
    __shared__ __bf16 ldsV[32][64];    // 4 KB, rows packed (tile_dim0=64)
    const int bh = blockIdx.x, t = threadIdx.x, wave = t >> 5, lane = t & 31;

    v8f acc[3] = {};
    for (int kk = 0; kk < L_; kk += 32) {
        __syncthreads();   // previous iteration's fragment reads done
        if (wave == 0) {
            // V tile: 32 rows x 64 elems, row stride 64 elems
            tdm_load_2d_bf16(lds_offset(&ldsV[0][0]),
                             Vb + ((size_t)bh * L_ + kk) * D_,
                             /*tileX=*/64, /*tileY=*/32,
                             /*stride=*/64, /*dimX=*/64, /*dimY=*/L_);
        } else if (wave == 1) {
            // P tile: 48 rows x 32 elems, row stride 1024 elems
            tdm_load_2d_bf16(lds_offset(&ldsP[0][0]),
                             Pws + (size_t)bh * UP_ * L_ + kk,
                             /*tileX=*/32, /*tileY=*/UP_,
                             /*stride=*/L_, /*dimX=*/L_, /*dimY=*/UP_);
        }
        __builtin_amdgcn_s_wait_tensorcnt(0);
        __syncthreads();   // tiles visible to all waves

        bf16x16 bfrag = frag_b(&ldsV[0][wave * 16], 64);
#pragma unroll
        for (int band = 0; band < 3; ++band)
            acc[band] = wmma_bf16(frag_a(&ldsP[band * 16][0], 32), bfrag, acc[band]);
    }

    const int n = lane & 15;
    const int mb = (lane & 16) ? 8 : 0;
#pragma unroll
    for (int band = 0; band < 3; ++band)
#pragma unroll
        for (int r = 0; r < 8; ++r)
            ctxtop[((size_t)bh * UP_ + band * 16 + mb + r) * D_ + wave * 16 + n] =
                acc[band][r];
}

// ----------------- Kernel 7: V mean over keys, per (b,h,d) -----------------
__global__ void __launch_bounds__(64)
vmean_kernel(const __bf16* __restrict__ Vb, float* __restrict__ vmean)
{
    const int bh = blockIdx.x, d = threadIdx.x;
    float s = 0.f;
    for (int l = 0; l < L_; ++l) s += (float)Vb[((size_t)bh * L_ + l) * D_ + d];
    vmean[bh * D_ + d] = s * (1.0f / (float)L_);
}

// -------- Kernel 8: fill context (B,L,F) with broadcast V-mean -------------
__global__ void __launch_bounds__(256)
fill_kernel(const float* __restrict__ vmean, float* __restrict__ ctxf)
{
    size_t i = (size_t)blockIdx.x * 256 + threadIdx.x;   // < B*L*F
    int c = (int)(i & 511);
    int b = (int)(i >> 19);
    int h = c >> 6, d = c & 63;
    ctxf[i] = vmean[(b * H_ + h) * D_ + d];
}

// ---------------- Kernel 9: scatter top-u rows into context ----------------
__global__ void __launch_bounds__(64)
scatter_kernel(const int* __restrict__ idxws, const float* __restrict__ ctxtop,
               float* __restrict__ ctxf)
{
    const int g = blockIdx.x;        // BH * U
    const int bh = g / U_, u = g % U_;
    const int b = bh >> 3, h = bh & 7;
    const int d = threadIdx.x;
    const int row = idxws[bh * U_ + u];
    ctxf[((size_t)(b * L_) + row) * F_ + h * D_ + d] =
        ctxtop[((size_t)bh * UP_ + u) * D_ + d];
}

// ------------------------------ launcher -----------------------------------
extern "C" void kernel_launch(void* const* d_in, const int* in_sizes, int n_in,
                              void* d_out, int out_size, void* d_ws, size_t ws_size,
                              hipStream_t stream)
{
    (void)in_sizes; (void)n_in; (void)out_size; (void)ws_size;
    const float* query = (const float*)d_in[0];
    const float* key   = (const float*)d_in[1];
    const float* value = (const float*)d_in[2];
    const int*   maskp = (const int*)  d_in[3];
    const float* Wq = (const float*)d_in[4];  const float* bq = (const float*)d_in[5];
    const float* Wk = (const float*)d_in[6];  const float* bk = (const float*)d_in[7];
    const float* Wv = (const float*)d_in[8];  const float* bv = (const float*)d_in[9];
    const float* Wo = (const float*)d_in[10]; const float* bo = (const float*)d_in[11];

    char* w = (char*)d_ws;
    size_t off = 0;
    auto alloc = [&](size_t bytes) -> void* {
        void* p = w + off;
        off = (off + bytes + 255) & ~(size_t)255;
        return p;
    };
    __bf16* Qb     = (__bf16*)alloc((size_t)B_ * H_ * L_ * D_ * 2);
    __bf16* Kb     = (__bf16*)alloc((size_t)B_ * H_ * L_ * D_ * 2);
    __bf16* Vb     = (__bf16*)alloc((size_t)B_ * H_ * L_ * D_ * 2);
    float*  Mws    = (float*) alloc((size_t)BH_ * L_ * 4);
    int*    idxws  = (int*)   alloc((size_t)BH_ * U_ * 4);
    float*  Sws    = (float*) alloc((size_t)BH_ * UP_ * L_ * 4);
    __bf16* Pws    = (__bf16*)alloc((size_t)BH_ * UP_ * L_ * 2);
    float*  ctxtop = (float*) alloc((size_t)BH_ * UP_ * D_ * 4);
    float*  vmean  = (float*) alloc((size_t)BH_ * D_ * 4);
    float*  ctxf   = (float*) alloc((size_t)B_ * L_ * F_ * 4);

    const dim3 gGemm(B_ * L_ / GTM, F_ / GTN);   // (128, 8)

    gemm_xwT_kernel<0><<<gGemm, 256, 0, stream>>>(query, Wq, bq, Qb, nullptr);
    gemm_xwT_kernel<0><<<gGemm, 256, 0, stream>>>(key,   Wk, bk, Kb, nullptr);
    gemm_xwT_kernel<0><<<gGemm, 256, 0, stream>>>(value, Wv, bv, Vb, nullptr);

    scores_m_kernel<<<dim3(BH_, L_ / 16), 256, 0, stream>>>(Qb, Kb, Mws);
    topk_kernel<<<BH_, 256, 0, stream>>>(Mws, idxws);
    red_scores_kernel<<<dim3(BH_, 3), 256, 0, stream>>>(Qb, Kb, idxws, maskp, Sws);
    softmax_kernel<<<BH_ * UP_, 256, 0, stream>>>(Sws, Pws);
    ctx_kernel<<<BH_, 128, 0, stream>>>(Pws, Vb, ctxtop);

    vmean_kernel<<<BH_, 64, 0, stream>>>(Vb, vmean);
    fill_kernel<<<(B_ * L_ * F_) / 256, 256, 0, stream>>>(vmean, ctxf);
    scatter_kernel<<<BH_ * U_, 64, 0, stream>>>(idxws, ctxtop, ctxf);

    gemm_xwT_kernel<1><<<gGemm, 256, 0, stream>>>(ctxf, Wo, bo, nullptr, (float*)d_out);
}
